// IIMA_26783416057972
// MI455X (gfx1250) — matrix operand order repaired and checked by reading, not compiled, not run on hardware
//
#include <hip/hip_runtime.h>
#include <math.h>

// Problem constants (from reference): [Q, N, C, H, W] = [200, 25, 640, 5, 5]
constexpr int kQ  = 200;
constexpr int kN  = 25;
constexpr int kC  = 640;
constexpr int kH  = 5;
constexpr int kW  = 5;
constexpr int kHW = kH * kW;          // 25 spatial positions
constexpr int kCHW = kC * kHW;        // 16000 elements per (q,n) tile
constexpr float kTempInv = 10.0f;     // 1 / 0.1
constexpr float kEps = 1e-8f;

typedef __attribute__((ext_vector_type(16))) _Float16 v16h;
typedef __attribute__((ext_vector_type(8)))  float    v8f;
typedef __attribute__((ext_vector_type(4)))  int      v4i;

// --- CDNA5 async global->LDS path (guarded; falls back to plain loads) -----
#if defined(__has_builtin)
#if __has_builtin(__builtin_amdgcn_global_load_async_to_lds_b128)
#define HAVE_ASYNC_LDS 1
#endif
#endif

#if defined(HAVE_ASYNC_LDS)
// From the compiler diagnostic: arg0 is 'int __vector(4) __device__ *'
// (16-byte vector in the global address space, non-const).
typedef __attribute__((address_space(1))) v4i* g128_t;
typedef __attribute__((address_space(3))) v4i* l128_t;
#if __has_builtin(__builtin_amdgcn_s_wait_asynccnt)
#define ASYNC_WAIT0() __builtin_amdgcn_s_wait_asynccnt(0)
#else
#define ASYNC_WAIT0() asm volatile("s_wait_asynccnt 0x0" ::: "memory")
#endif
#endif

// ---------------------------------------------------------------------------
// Zero the workspace accumulators (attended_s sum + q_att sum).
// ---------------------------------------------------------------------------
__global__ void IIMA_zero_kernel(float* __restrict__ p, int n) {
  int i = blockIdx.x * blockDim.x + threadIdx.x;
  if (i < n) p[i] = 0.0f;
}

// ---------------------------------------------------------------------------
// Per-(q,n) block: async-stage feature_s tile to LDS, correlation GEMM via
// WMMA (fp16 in / fp32 acc), 5-wide softmaxes, atomic Q-mean accumulation.
// ---------------------------------------------------------------------------
__global__ __launch_bounds__(256) void IIMA_corr_attn_kernel(
    const float* __restrict__ fq, const float* __restrict__ fs,
    float* __restrict__ attSum /* [N][C][HW] */,
    float* __restrict__ qAttSum /* [N][HW] */) {
  extern __shared__ char smem_raw[];
  float*    Sf  = reinterpret_cast<float*>(smem_raw);      // [16000] raw f32 S tile
  _Float16* St  = reinterpret_cast<_Float16*>(Sf + kCHW);  // [32][640] f16 (rows 25..31 = 0)
  _Float16* Qt  = St + 32 * kC;                            // [32][640] f16
  float* corr   = reinterpret_cast<float*>(Qt + 32 * kC);  // [32][33] (stride 33 vs bank clash)
  float* m_s    = corr + 32 * 33;                          // 32
  float* m_q    = m_s + 32;                                // 32
  float* s_att  = m_q + 32;                                // 32
  float* q_att  = s_att + 32;                              // 32

  const int bid = blockIdx.x;          // bid = q*kN + n
  const int n   = bid % kN;
  const int tid = threadIdx.x;

  const float* Sg = fs + (size_t)bid * kCHW;   // feature_s tile [c][hw], hw fastest
  const float* Qg = fq + (size_t)bid * kCHW;   // feature_q tile

  // ---- Stage raw feature_s tile (64000 B) into LDS ----
#if defined(HAVE_ASYNC_LDS)
  {
    // Async DMA: 4000 b128 transfers, no VGPR round-trip (tracked by ASYNCcnt).
    g128_t g = (g128_t)Sg;
    l128_t l = (l128_t)Sf;
    const int nvec = kCHW * 4 / 16;    // 4000 16-byte transfers
    for (int i = tid; i < nvec; i += 256)
      __builtin_amdgcn_global_load_async_to_lds_b128(g + i, l + i, 0, 0);
    ASYNC_WAIT0();
  }
#else
  for (int e = tid; e < kCHW; e += 256) Sf[e] = Sg[e];
#endif
  __syncthreads();

  // ---- Transpose + fp16 convert into WMMA operand tiles ----
  // (These values only feed attention logits -> softmax, fp16 is plenty.)
  for (int e = tid; e < kCHW; e += 256) {
    const int c = e / kHW, hw = e % kHW;
    St[hw * kC + c] = (_Float16)Sf[e];
    Qt[hw * kC + c] = (_Float16)Qg[e];
  }
  for (int e = tid; e < 7 * kC; e += 256) {  // zero pad rows/cols 25..31
    St[25 * kC + e] = (_Float16)0.0f;
    Qt[25 * kC + e] = (_Float16)0.0f;
  }
  __syncthreads();

  // corr[ij][kl] = sum_c S[c][ij] * Q[c][kl]   (M=N=32 padded, K=640)
  // 4 waves each own one 16x16 tile; 20 k-steps of v_wmma_f32_16x16x32_f16.
  const int wave = tid >> 5;
  const int lane = tid & 31;
  if (wave < 4) {
    const int MB = (wave >> 1) << 4;   // A-tile row base (s spatial)
    const int NB = (wave & 1) << 4;    // B-tile col base (q spatial)
    const int half = lane >> 4;        // ISA 16-bit fragment lane halves
    const int l    = lane & 15;
    const _Float16* arow = St + (MB + l) * kC;  // A[M][K]: row M = s position
    const _Float16* brow = Qt + (NB + l) * kC;  // B[K][N]: lane l holds column N
    v8f acc = {};
    for (int k0 = 0; k0 < kC; k0 += 32) {
      union { v16h v; _Float16 h[16]; } A, B;
      // A 16x32 f16 layout: half0 -> K[k0..k0+7],[k0+16..k0+23]; half1 -> +8
      const _Float16* a1 = arow + k0 + half * 8;
      const _Float16* a2 = arow + k0 + 16 + half * 8;
#pragma unroll
      for (int i = 0; i < 8; ++i) { A.h[i] = a1[i]; A.h[8 + i] = a2[i]; }
      // B 32x16 f16 layout: half0 -> K[k0..k0+7],[k0+16..k0+23]; half1 -> +8
      const _Float16* b1 = brow + k0 + half * 8;
      const _Float16* b2 = brow + k0 + 16 + half * 8;
#pragma unroll
      for (int i = 0; i < 8; ++i) { B.h[i] = b1[i]; B.h[8 + i] = b2[i]; }
      acc = __builtin_amdgcn_wmma_f32_16x16x32_f16(
          /*neg_a=*/false, A.v, /*neg_b=*/false, B.v,
          /*c_mod=*/(short)0, acc, /*reuse_a=*/false, /*reuse_b=*/false);
    }
    // D layout: VGPR r, lanes 0-15 -> M=r, lanes 16-31 -> M=r+8; N = lane&15
#pragma unroll
    for (int r = 0; r < 8; ++r)
      corr[(MB + r + 8 * half) * 33 + (NB + l)] = acc[r];
  }
  __syncthreads();

  // Row mean -> m_s[ij]; column mean -> m_q[kl]. Pad entries are exact zeros.
  if (tid < kHW) {
    float rs = 0.0f, cs = 0.0f;
#pragma unroll
    for (int j = 0; j < 32; ++j) {
      rs += corr[tid * 33 + j];
      cs += corr[j * 33 + tid];
    }
    m_s[tid] = rs * (1.0f / (float)kHW);
    m_q[tid] = cs * (1.0f / (float)kHW);
  }
  __syncthreads();

  // softmax over the LAST axis only (W = 5) per spatial row, torch semantics.
  if (tid < kH) {                       // s_att row tid
    const float* src = m_s + tid * kW;
    float* dst = s_att + tid * kW;
    float mx = src[0];
#pragma unroll
    for (int i = 1; i < kW; ++i) mx = fmaxf(mx, src[i]);
    float sum = 0.0f;
#pragma unroll
    for (int i = 0; i < kW; ++i) { float e = __expf(src[i] - mx); dst[i] = e; sum += e; }
    const float inv = 1.0f / sum;
#pragma unroll
    for (int i = 0; i < kW; ++i) dst[i] *= inv;
  } else if (tid >= 8 && tid < 8 + kH) {  // q_att row tid-8
    const int r = tid - 8;
    const float* src = m_q + r * kW;
    float* dst = q_att + r * kW;
    float mx = src[0];
#pragma unroll
    for (int i = 1; i < kW; ++i) mx = fmaxf(mx, src[i]);
    float sum = 0.0f;
#pragma unroll
    for (int i = 0; i < kW; ++i) { float e = __expf(src[i] - mx); dst[i] = e; sum += e; }
    const float inv = 1.0f / sum;
#pragma unroll
    for (int i = 0; i < kW; ++i) dst[i] *= inv;
  }
  __syncthreads();

  // Accumulate Q-mean reductions (buffers are L2-resident: 1.6 MB total).
  if (tid < kHW) atomicAdd(qAttSum + n * kHW + tid, q_att[tid]);

  const float invQ = 1.0f / (float)kQ;
  float* aBase = attSum + n * kCHW;
  for (int e = tid; e < kCHW; e += 256) {
    const int hw = e % kHW;
    atomicAdd(aBase + e, s_att[hw] * Sf[e] * invQ);  // S read from LDS, no re-fetch
  }
}

// ---------------------------------------------------------------------------
// Finalize: cosine similarity along W with eps clamp, /temperature.
// One thread per output element (q,n,c,h); feature_q reads fully coalesced.
// ---------------------------------------------------------------------------
__global__ __launch_bounds__(256) void IIMA_finalize_kernel(
    const float* __restrict__ fq, const float* __restrict__ attSum,
    const float* __restrict__ qAttSum, float* __restrict__ out) {
  const int total = kQ * kN * kC * kH;     // 16,000,000
  const int i = blockIdx.x * 256 + threadIdx.x;
  if (i >= total) return;

  const int h = i % kH;
  const int c = (i / kH) % kC;
  const int n = (i / (kH * kC)) % kN;

  const float invQ = 1.0f / (float)kQ;
  const float* fqp = fq + (size_t)i * kW;                           // [q][n][c][h][:]
  const float* asp = attSum + (size_t)(n * kC + c) * kHW + h * kW;  // already /Q
  const float* qsp = qAttSum + n * kHW + h * kW;

  float dot = 0.0f, ss = 0.0f, qq = 0.0f;
#pragma unroll
  for (int w = 0; w < kW; ++w) {
    const float a = asp[w];
    const float b = fqp[w] * (qsp[w] * invQ);
    dot += a * b;
    ss  += a * a;
    qq  += b * b;
  }
  const float ns = fmaxf(sqrtf(ss), kEps);
  const float nq = fmaxf(sqrtf(qq), kEps);
  out[i] = dot / (ns * nq) * kTempInv;
}

// ---------------------------------------------------------------------------
extern "C" void kernel_launch(void* const* d_in, const int* in_sizes, int n_in,
                              void* d_out, int out_size, void* d_ws, size_t ws_size,
                              hipStream_t stream) {
  const float* fq = (const float*)d_in[0];   // feature_q [Q,N,C,H,W] fp32
  const float* fs = (const float*)d_in[1];   // feature_s [Q,N,C,H,W] fp32
  float* out = (float*)d_out;                // [Q,N,C,H] fp32

  float* attSum  = (float*)d_ws;             // [N][C][HW] = 400,000 f32
  float* qAttSum = attSum + kN * kCHW;       // [N][HW]    = 625 f32

  // 1) zero workspace accumulators
  const int nz = kN * kCHW + kN * kHW;
  IIMA_zero_kernel<<<(nz + 255) / 256, 256, 0, stream>>>(attSum, nz);

  // 2) correlation + attention + Q-mean accumulation (one block per (q,n))
  const size_t smem = (size_t)kCHW * sizeof(float)                 // Sf (raw f32)
                    + (size_t)(2 * 32 * kC) * sizeof(_Float16)     // St + Qt
                    + (size_t)(32 * 33 + 4 * 32) * sizeof(float);  // corr + vecs
  (void)hipFuncSetAttribute(
      reinterpret_cast<const void*>(IIMA_corr_attn_kernel),
      hipFuncAttributeMaxDynamicSharedMemorySize, (int)smem);
  IIMA_corr_attn_kernel<<<kQ * kN, 256, smem, stream>>>(fq, fs, attSum, qAttSum);

  // 3) cosine similarity finalize
  const int total = kQ * kN * kC * kH;
  IIMA_finalize_kernel<<<(total + 255) / 256, 256, 0, stream>>>(fq, attSum,
                                                                qAttSum, out);
}
